// SelfAttention_19473381720102
// MI455X (gfx1250) — matrix operand order repaired and checked by reading, not compiled, not run on hardware
//
#include <hip/hip_runtime.h>
#include <hip/hip_bf16.h>

typedef __attribute__((ext_vector_type(16))) __bf16 v16bf;
typedef __attribute__((ext_vector_type(8)))  __bf16 bf16x8;
typedef __attribute__((ext_vector_type(8)))  float  v8f;
typedef __attribute__((ext_vector_type(4)))  unsigned int u32x4;
typedef __attribute__((ext_vector_type(8)))  int i32x8;
typedef __attribute__((ext_vector_type(4)))  int i32x4;

#if defined(__has_builtin)
#if __has_builtin(__builtin_amdgcn_tensor_load_to_lds) && \
    __has_builtin(__builtin_amdgcn_s_wait_tensorcnt)
#define USE_TDM 1
#endif
#endif
#ifndef USE_TDM
#define USE_TDM 0
#endif

// ---------------------------------------------------------------------------
// WMMA fragment loaders for v_wmma_f32_16x16x32_bf16 (wave32).
// A (16x32, MxK): lanes 0-15: M=lane, K=0..15; lanes 16-31: M=lane-16,
//   K=16..31.
// B (32x16, KxN): lanes 0-15: N=lane, K=0..15; lanes 16-31: N=lane-16,
//   K=16..31; element j <-> K = khalf + j.
// C/D (16x16 f32): vgpr r: lanes 0-15 -> M=r, lanes 16-31 -> M=r+8; N=lane%16.
// Works for global or LDS base pointers (LDS resolves to ds_load_b128).
// ---------------------------------------------------------------------------
__device__ __forceinline__ v16bf load_frag_a(const __bf16* base, int ld,
                                             int m0, int k0) {
    int lane = threadIdx.x & 31;
    int m    = m0 + (lane & 15);
    int half = (lane >> 4) & 1;
    const __bf16* p = base + (long)m * ld + k0 + half * 16;
    bf16x8 lo = *(const bf16x8*)(p);
    bf16x8 hi = *(const bf16x8*)(p + 8);
    v16bf r;
#pragma unroll
    for (int j = 0; j < 8; ++j) { r[j] = lo[j]; r[j + 8] = hi[j]; }
    return r;
}

__device__ __forceinline__ v16bf load_frag_b(const __bf16* base, int ld,
                                             int n0, int k0) {
    int lane  = threadIdx.x & 31;
    int n     = n0 + (lane & 15);
    int khalf = (lane >> 4) ? 16 : 0;
    const __bf16* p = base + (long)n * ld + k0 + khalf;
    bf16x8 lo = *(const bf16x8*)(p);
    bf16x8 hi = *(const bf16x8*)(p + 8);
    v16bf r;
#pragma unroll
    for (int j = 0; j < 8; ++j) { r[j] = lo[j]; r[j + 8] = hi[j]; }
    return r;
}

__device__ __forceinline__ v8f wmma_bf16(v16bf a, v16bf b, v8f c) {
    return __builtin_amdgcn_wmma_f32_16x16x32_bf16(false, a, false, b,
                                                   (short)0, c, false, false);
}

#if USE_TDM
// ---------------------------------------------------------------------------
// Tensor Data Mover: build D# groups per ISA 8.3/8.4 and issue a 2D tile load
// (tile1 rows x tile0 contiguous elements) of bf16 data into LDS.
// Group0: [1:0]=count=1, [63:32]=lds_addr, [120:64]=global_addr, [127:126]=2.
// Group1: data_size=1 (2B), tensor_dim0/1, tile_dim0/1, tensor_dim0_stride.
// This toolchain exposes the 6-arg builtin:
//   (u32x4 g0, i32x8 g1, i32x4 g2, i32x4 g3, i32x8 extra, i32 cpol)
// ---------------------------------------------------------------------------
__device__ __forceinline__ void tdm_load_2d(unsigned lds_addr,
                                            unsigned long long gaddr,
                                            unsigned td0, unsigned td1,
                                            unsigned tile0, unsigned tile1,
                                            unsigned long long stride0) {
    u32x4 g0;
    g0[0] = 1u;                                   // count=1, no gather
    g0[1] = lds_addr;                             // LDS byte address
    g0[2] = (unsigned)(gaddr & 0xffffffffull);    // global addr [31:0]
    g0[3] = (unsigned)((gaddr >> 32) & 0x01ffffffull) | 0x80000000u; // +type=2
    i32x8 g1;
    g1[0] = (int)(1u << 16);                      // data_size = 2 bytes
    g1[1] = (int)((td0 & 0xffffu) << 16);
    g1[2] = (int)(((td0 >> 16) & 0xffffu) | ((td1 & 0xffffu) << 16));
    g1[3] = (int)(((td1 >> 16) & 0xffffu) | ((tile0 & 0xffffu) << 16));
    g1[4] = (int)(tile1 & 0xffffu);               // tile_dim2 = 0
    g1[5] = (int)(stride0 & 0xffffffffull);
    g1[6] = (int)((stride0 >> 32) & 0xffffull);   // dim1_stride = 0
    g1[7] = 0;
    i32x4 z4 = {};
    i32x8 z8 = {};
    __builtin_amdgcn_tensor_load_to_lds(g0, g1, z4, z4, z8, 0);
}
#endif

// ---------------------------------------------------------------------------
// Elementwise f32 -> bf16 convert (weights)
// ---------------------------------------------------------------------------
__global__ void cvt_f32_bf16(const float* __restrict__ in,
                             __bf16* __restrict__ out, int n) {
    int i = blockIdx.x * 256 + threadIdx.x;
    if (i < n) out[i] = (__bf16)in[i];
}

// ---------------------------------------------------------------------------
// Tiled transpose: out[n*M + m] = (bf16) in[m*N + n], per-batch (blockIdx.z).
// ---------------------------------------------------------------------------
template <typename T>
__global__ void transpose_kernel(const T* __restrict__ in,
                                 __bf16* __restrict__ out, int M, int N) {
    __shared__ __bf16 tile[32][33];
    long boff = (long)blockIdx.z * M * N;
    const T* I = in + boff;
    __bf16*  O = out + boff;
    int m0 = blockIdx.y * 32, n0 = blockIdx.x * 32;
    int tx = threadIdx.x, ty = threadIdx.y;   // block (32, 8)
#pragma unroll
    for (int r = 0; r < 32; r += 8)
        tile[ty + r][tx] = (__bf16)I[(long)(m0 + ty + r) * N + n0 + tx];
    __syncthreads();
#pragma unroll
    for (int r = 0; r < 32; r += 8)
        O[(long)(n0 + ty + r) * M + m0 + tx] = tile[tx][ty + r];
}

// ---------------------------------------------------------------------------
// WMMA GEMM:  C[M x N] = A[M x K] * B (B supplied N-major: N x K row-major).
// Software-pipelined k-loop: fragments for k+32 are fetched while the k-step
// WMMAs issue, so s_wait_loadcnt only covers the older group.
// Block 256 (8 waves); wave computes 16(M) x 64(N). Grid (N/256, M/32, B).
// ---------------------------------------------------------------------------
__global__ __launch_bounds__(256) void gemm_wmma(
    const __bf16* __restrict__ A, const __bf16* __restrict__ Bs,
    __bf16* __restrict__ Cb, float* __restrict__ Cf, int M, int N, int K) {
    int w    = threadIdx.x >> 5;
    int lane = threadIdx.x & 31;
    int m0 = blockIdx.y * 32 + (w >> 2) * 16;
    int n0 = blockIdx.x * 256 + (w & 3) * 64;
    const __bf16* B = Bs + (long)blockIdx.z * N * K;
    long coff = (long)blockIdx.z * M * N;

    v8f acc[4] = {};
    v16bf ac = load_frag_a(A, K, m0, 0);
    v16bf bc[4];
#pragma unroll
    for (int t = 0; t < 4; ++t) bc[t] = load_frag_b(B, K, n0 + 16 * t, 0);

    for (int k = 0; k < K; k += 32) {
        int kn = (k + 32 < K) ? (k + 32) : k;   // last iter: harmless re-fetch
        v16bf an = load_frag_a(A, K, m0, kn);
        v16bf bn[4];
#pragma unroll
        for (int t = 0; t < 4; ++t) bn[t] = load_frag_b(B, K, n0 + 16 * t, kn);
#pragma unroll
        for (int t = 0; t < 4; ++t) acc[t] = wmma_bf16(ac, bc[t], acc[t]);
        ac = an;
#pragma unroll
        for (int t = 0; t < 4; ++t) bc[t] = bn[t];
    }

    int n  = (lane & 15);
    int mo = m0 + ((lane >> 4) ? 8 : 0);
    if (Cf) {
        float* C = Cf + coff;
#pragma unroll
        for (int t = 0; t < 4; ++t)
#pragma unroll
            for (int r = 0; r < 8; ++r)
                C[(long)(mo + r) * N + n0 + 16 * t + n] = acc[t][r];
    } else {
        __bf16* C = Cb + coff;
#pragma unroll
        for (int t = 0; t < 4; ++t)
#pragma unroll
            for (int r = 0; r < 8; ++r)
                C[(long)(mo + r) * N + n0 + 16 * t + n] = (__bf16)acc[t][r];
    }
}

// ---------------------------------------------------------------------------
// Fused flash attention (per batch). HW = 4096, C = 256, scale = 1/16.
// K/V chunks (32 keys) are staged into LDS once per block — via the Tensor
// Data Mover (double-buffered, TENSORcnt-synchronized) when available — and
// all 8 waves compute their WMMA fragments from LDS (ds_load_b128).
//   S^T[t,i] = sum_c Kt[t,c] * q[i,c]   (A = LDS K rows, B = q rows)
//   O^T[c,i] = sum_t Vt[c,t] * P^T[t,i] (A = LDS V rows, B = P built in-reg)
// Grid (HW/128, B), block 256 (8 waves, 16 queries each).
// ---------------------------------------------------------------------------
__global__ __launch_bounds__(256, 1) void attn_wmma(
    const __bf16* __restrict__ Q,   // per batch: 4096 x 256 (q rows, K-contig)
    const __bf16* __restrict__ Kt,  // per batch: 4096 x 256
    const __bf16* __restrict__ Vt,  // per batch: 256 x 4096
    __bf16* __restrict__ G) {       // per batch: 4096 x 256
    const int HW = 4096, C = 256, NCH = HW / 32;
#if USE_TDM
    __shared__ __bf16 lk[2][32 * 256];
    __shared__ __bf16 lv[2][256 * 32];
#else
    __shared__ __bf16 lk[1][32 * 256];
    __shared__ __bf16 lv[1][256 * 32];
#endif
    int w    = threadIdx.x >> 5;
    int lane = threadIdx.x & 31;
    long bo  = (long)blockIdx.y * HW * C;
    const __bf16* q  = Q + bo;
    const __bf16* kt = Kt + bo;
    const __bf16* vt = Vt + bo;
    __bf16* g = G + bo;
    int q0 = blockIdx.x * 128 + w * 16;

    // Preload this wave's 8 q B-fragments (reused for every key chunk).
    v16bf qb[8];
#pragma unroll
    for (int ks = 0; ks < 8; ++ks) qb[ks] = load_frag_b(q, C, q0, ks * 32);

    v8f o[16] = {};
    float mrun = -3.0e38f, lrun = 0.0f;
    const float sc = 0.0625f;       // 1/sqrt(256)

#if USE_TDM
    unsigned lk_a[2], lv_a[2];
#pragma unroll
    for (int p = 0; p < 2; ++p) {
        lk_a[p] = (unsigned)(unsigned long long)&lk[p][0];
        lv_a[p] = (unsigned)(unsigned long long)&lv[p][0];
    }
    if (threadIdx.x < 32) {  // wave 0 drives the TDM
        // chunk 0: K rows [0,32) are contiguous 8192 elems; V is 256x32 strided
        tdm_load_2d(lk_a[0], (unsigned long long)kt, 8192u, 1u, 8192u, 1u, 8192u);
        tdm_load_2d(lv_a[0], (unsigned long long)vt, 4096u, 256u, 32u, 256u, 4096u);
    }
#endif

    for (int ic = 0; ic < NCH; ++ic) {
        int kc = ic * 32;
#if USE_TDM
        int cur = ic & 1;
        if (threadIdx.x < 32) {
            if (ic + 1 < NCH) {
                int kn = kc + 32;
                tdm_load_2d(lk_a[cur ^ 1],
                            (unsigned long long)(kt + (long)kn * 256),
                            8192u, 1u, 8192u, 1u, 8192u);
                tdm_load_2d(lv_a[cur ^ 1],
                            (unsigned long long)(vt + kn),
                            4096u, 256u, 32u, 256u, 4096u);
                __builtin_amdgcn_s_wait_tensorcnt(2);  // chunk ic resident
            } else {
                __builtin_amdgcn_s_wait_tensorcnt(0);
            }
        }
        __syncthreads();
#else
        int cur = 0;
        {   // cooperative staging fallback: 2 x 16KB as b128 copies
            const uint4* ks_ = (const uint4*)(kt + (long)kc * 256);
            uint4* kd = (uint4*)&lk[0][0];
            for (int i = threadIdx.x; i < 1024; i += 256) kd[i] = ks_[i];
            uint4* vd = (uint4*)&lv[0][0];
            for (int i = threadIdx.x; i < 1024; i += 256) {
                int r = i >> 2, cpart = i & 3;
                vd[i] = *(const uint4*)(vt + (long)r * HW + kc + cpart * 8);
            }
        }
        __syncthreads();
#endif
        const __bf16* kl = &lk[cur][0];
        const __bf16* vl = &lv[cur][0];

        // ---- scores S^T for 32 keys x 16 queries (from LDS) ----
        v8f s0 = {}, s1 = {};
#pragma unroll
        for (int ks = 0; ks < 8; ++ks) {
            v16bf a0 = load_frag_a(kl, C, 0, ks * 32);
            v16bf a1 = load_frag_a(kl, C, 16, ks * 32);
            s0 = wmma_bf16(a0, qb[ks], s0);
            s1 = wmma_bf16(a1, qb[ks], s1);
        }
        // ---- online softmax over key dim (in-lane vgprs + half swap) ----
        float cm = -3.0e38f;
#pragma unroll
        for (int r = 0; r < 8; ++r) {
            s0[r] *= sc; s1[r] *= sc;
            cm = fmaxf(cm, fmaxf(s0[r], s1[r]));
        }
        cm = fmaxf(cm, __shfl_xor(cm, 16, 32));
        float mnew  = fmaxf(mrun, cm);
        float alpha = __expf(mrun - mnew);
        float p0[8], p1[8];
        float rsum = 0.0f;
#pragma unroll
        for (int r = 0; r < 8; ++r) {
            p0[r] = __expf(s0[r] - mnew);
            p1[r] = __expf(s1[r] - mnew);
            rsum += p0[r] + p1[r];
        }
        rsum += __shfl_xor(rsum, 16, 32);
        lrun = lrun * alpha + rsum;
        mrun = mnew;
#pragma unroll
        for (int t = 0; t < 16; ++t)
#pragma unroll
            for (int r = 0; r < 8; ++r) o[t][r] *= alpha;

        // ---- repack P (S^T D-layout) into a 32x16 B-fragment ----
        float p0x[8], p1x[8];
#pragma unroll
        for (int r = 0; r < 8; ++r) {
            p0x[r] = __shfl_xor(p0[r], 16, 32);
            p1x[r] = __shfl_xor(p1[r], 16, 32);
        }
        bool lo = (lane < 16);
        v16bf pb;
#pragma unroll
        for (int j = 0; j < 8; ++j) pb[j]     = (__bf16)(lo ? p0[j]  : p1x[j]);
#pragma unroll
        for (int j = 0; j < 8; ++j) pb[j + 8] = (__bf16)(lo ? p0x[j] : p1[j]);

        // ---- O^T += V-tiles @ P (A from LDS, ld = 32) ----
#pragma unroll
        for (int t = 0; t < 16; ++t) {
            v16bf av = load_frag_a(vl, 32, t * 16, 0);
            o[t] = wmma_bf16(av, pb, o[t]);
        }
        __syncthreads();
    }

    float inv = 1.0f / lrun;
    int half8 = (lane >> 4) * 8;
    int qi    = q0 + (lane & 15);
#pragma unroll
    for (int t = 0; t < 16; ++t)
#pragma unroll
        for (int r = 0; r < 8; ++r)
            g[(long)qi * C + t * 16 + r + half8] = (__bf16)(o[t][r] * inv);
}

// ---------------------------------------------------------------------------
// Host-side launch.
// ---------------------------------------------------------------------------
extern "C" void kernel_launch(void* const* d_in, const int* in_sizes, int n_in,
                              void* d_out, int out_size, void* d_ws,
                              size_t ws_size, hipStream_t stream) {
    const int B = 8, C = 256, HW = 4096;
    const float* X  = (const float*)d_in[0];
    const float* Wq = (const float*)d_in[1];
    const float* Wk = (const float*)d_in[2];
    const float* Wv = (const float*)d_in[3];
    const float* Wp = (const float*)d_in[4];
    float* out = (float*)d_out;

    const size_t NB = (size_t)B * C * HW * sizeof(__bf16);  // 16 MB
    char* ws = (char*)d_ws;
    __bf16* Xt  = (__bf16*)(ws);
    __bf16* Qb  = (__bf16*)(ws + 1 * NB);
    __bf16* Kc  = (__bf16*)(ws + 2 * NB);
    __bf16* Ktp = (__bf16*)(ws + 3 * NB);
    __bf16* Vc  = (__bf16*)(ws + 4 * NB);
    __bf16* Vtp = (__bf16*)(ws + 5 * NB);
    __bf16* G   = Kc;   // reuse
    __bf16* Gvt = Vc;   // reuse
    __bf16* Wqb = (__bf16*)(ws + 6 * NB);
    __bf16* Wkb = Wqb + C * C;
    __bf16* Wvb = Wkb + C * C;
    __bf16* Wpb = Wvb + C * C;

    dim3 tb(32, 8);
    cvt_f32_bf16<<<dim3((C * C + 255) / 256), 256, 0, stream>>>(Wq, Wqb, C * C);
    cvt_f32_bf16<<<dim3((C * C + 255) / 256), 256, 0, stream>>>(Wk, Wkb, C * C);
    cvt_f32_bf16<<<dim3((C * C + 255) / 256), 256, 0, stream>>>(Wv, Wvb, C * C);
    cvt_f32_bf16<<<dim3((C * C + 255) / 256), 256, 0, stream>>>(Wp, Wpb, C * C);

    transpose_kernel<float><<<dim3(HW / 32, C / 32, B), tb, 0, stream>>>(
        X, Xt, C, HW);

    dim3 gg(HW / 256, C / 32, B);
    gemm_wmma<<<gg, 256, 0, stream>>>(Wqb, Xt, Qb, nullptr, C, HW, C);
    gemm_wmma<<<gg, 256, 0, stream>>>(Wkb, Xt, Kc, nullptr, C, HW, C);
    gemm_wmma<<<gg, 256, 0, stream>>>(Wvb, Xt, Vc, nullptr, C, HW, C);

    transpose_kernel<__bf16><<<dim3(HW / 32, C / 32, B), tb, 0, stream>>>(
        Kc, Ktp, C, HW);
    transpose_kernel<__bf16><<<dim3(C / 32, HW / 32, B), tb, 0, stream>>>(
        Vc, Vtp, HW, C);

    attn_wmma<<<dim3(HW / 128, B), 256, 0, stream>>>(Qb, Ktp, Vtp, G);

    transpose_kernel<__bf16><<<dim3(HW / 32, C / 32, B), tb, 0, stream>>>(
        G, Gvt, C, HW);
    gemm_wmma<<<gg, 256, 0, stream>>>(Wpb, Gvt, nullptr, out, C, HW, C);
}
